// SelfAttention2D_85134841742061
// MI455X (gfx1250) — compile-verified, matrix operand order checked by reading, not run on hardware
//
#include <hip/hip_runtime.h>

#define BB 8
#define CC 256
#define NN 4096
#define DRR 32

typedef __attribute__((ext_vector_type(16))) __bf16 bf16x16;
typedef __attribute__((ext_vector_type(8)))  float  f32x8;

union BfOp { unsigned u[8]; bf16x16 v; };

// ---- bf16 pack: single hardware op -----------------------------------------
#if __has_builtin(__builtin_amdgcn_cvt_pk_bf16_f32)
__device__ __forceinline__ unsigned pack2bf(float lo, float hi) {
  auto r = __builtin_amdgcn_cvt_pk_bf16_f32(lo, hi);  // v_cvt_pk_bf16_f32 (RNE)
  return __builtin_bit_cast(unsigned, r);
}
#else
__device__ __forceinline__ unsigned pack2bf(float lo, float hi) {
  // v_perm_b32: take high halves of {hi, lo} -> packed bf16 (truncate)
  return __builtin_amdgcn_perm(__builtin_bit_cast(unsigned, hi),
                               __builtin_bit_cast(unsigned, lo), 0x07060302u);
}
#endif
__device__ __forceinline__ unsigned short f2bf(float f) {
  return (unsigned short)(pack2bf(f, 0.f) & 0xFFFFu);
}

// ---- load a 16x32 bf16 WMMA operand whose K-pairs are contiguous -----------
// Per ISA layout: u[0..3] = K {0..7}+8*hi ; u[4..7] = K {16..23}+8*hi
// -> two 16B-aligned b128 loads per operand.
__device__ __forceinline__ BfOp load_op_b128(const unsigned short* row, int hi) {
  BfOp o;
  uint4 a = *(const uint4*)(row + 8 * hi);
  uint4 b = *(const uint4*)(row + 16 + 8 * hi);
  o.u[0] = a.x; o.u[1] = a.y; o.u[2] = a.z; o.u[3] = a.w;
  o.u[4] = b.x; o.u[5] = b.y; o.u[6] = b.z; o.u[7] = b.w;
  return o;
}

// ---------------------------------------------------------------------------
// Kernel 1: Q/K/V 1x1-conv projections as bf16 WMMA GEMMs.
// Q,K -> [B,N,32] bf16 ; V -> [B,C,N] bf16.
// One wave per (b, 16-row i-tile); A panel (x^T 16x256) resident in VGPRs.
// ---------------------------------------------------------------------------
__global__ __launch_bounds__(128) void qkv_proj_kernel(
    const float* __restrict__ x,
    const float* __restrict__ Wq, const float* __restrict__ bq,
    const float* __restrict__ Wk, const float* __restrict__ bk,
    const float* __restrict__ Wv, const float* __restrict__ bv,
    unsigned short* __restrict__ Qws, unsigned short* __restrict__ Kws,
    unsigned short* __restrict__ Vws) {
  const int lane = threadIdx.x & 31;
  const int wave = threadIdx.x >> 5;
  const int w    = blockIdx.x * 4 + wave;
  const int b    = w / (NN / 16);
  const int i0   = (w % (NN / 16)) * 16;
  const int mrow = lane & 15;
  const int hi   = (lane >= 16) ? 1 : 0;

  // A panel: x^T tile [16 i x 256 c], 8 k-steps; pairs run along c (stride N)
  BfOp aP[8];
  #pragma unroll
  for (int ks = 0; ks < 8; ++ks) {
    #pragma unroll
    for (int v = 0; v < 8; ++v) {
      int c = ks * 32 + ((v < 4) ? 2 * v : 16 + 2 * (v - 4)) + 8 * hi;
      const float* p = x + (((size_t)b * CC + c) * NN + i0 + mrow);
      aP[ks].u[v] = pack2bf(p[0], p[NN]);
    }
  }

  #pragma unroll 1
  for (int nt = 0; nt < 20; ++nt) {
    const float* Wm; const float* bias; int dbase;
    if (nt < 2)      { Wm = Wq; bias = bq; dbase = nt * 16; }
    else if (nt < 4) { Wm = Wk; bias = bk; dbase = (nt - 2) * 16; }
    else             { Wm = Wv; bias = bv; dbase = (nt - 4) * 16; }
    const int d = dbase + mrow;

    f32x8 acc = {};
    #pragma unroll
    for (int ks = 0; ks < 8; ++ks) {
      // B operand: W^T pairs contiguous along c -> 4x float4 loads + 8 packs
      const float* wr = Wm + (size_t)d * CC + ks * 32 + 8 * hi;
      float4 wa = *(const float4*)(wr);
      float4 wb = *(const float4*)(wr + 4);
      float4 wc = *(const float4*)(wr + 16);
      float4 wd = *(const float4*)(wr + 20);
      BfOp bop;
      bop.u[0] = pack2bf(wa.x, wa.y); bop.u[1] = pack2bf(wa.z, wa.w);
      bop.u[2] = pack2bf(wb.x, wb.y); bop.u[3] = pack2bf(wb.z, wb.w);
      bop.u[4] = pack2bf(wc.x, wc.y); bop.u[5] = pack2bf(wc.z, wc.w);
      bop.u[6] = pack2bf(wd.x, wd.y); bop.u[7] = pack2bf(wd.z, wd.w);
      acc = __builtin_amdgcn_wmma_f32_16x16x32_bf16(
          false, aP[ks].v, false, bop.v, (short)0, acc, false, false);
    }

    const float bval = bias[d];
    if (nt < 4) {
      unsigned short* dst = (nt < 2) ? Qws : Kws;
      #pragma unroll
      for (int r = 0; r < 8; ++r) {
        int i = i0 + r + 8 * hi;
        dst[((size_t)b * NN + i) * DRR + d] = f2bf(acc[r] + bval);
      }
    } else {
      // V store: i runs contiguously with r -> one b128 store of 8 bf16
      uint4 pk;
      pk.x = pack2bf(acc[0] + bval, acc[1] + bval);
      pk.y = pack2bf(acc[2] + bval, acc[3] + bval);
      pk.z = pack2bf(acc[4] + bval, acc[5] + bval);
      pk.w = pack2bf(acc[6] + bval, acc[7] + bval);
      *(uint4*)(Vws + ((size_t)b * CC + d) * NN + i0 + 8 * hi) = pk;
    }
  }
}

// ---------------------------------------------------------------------------
// Kernel 2: flash-attention. One wave per (b, 16-row i-tile, 64-channel
// quarter). j streamed in blocks of 64: 4 WMMAs for S, one online-softmax
// update, P transposed via wave-private LDS slice, 8 WMMAs for O += P*V^T.
// Final: y = gamma * (O / l) + x.
// ---------------------------------------------------------------------------
__global__ __launch_bounds__(256) void attn_kernel(
    const float* __restrict__ x,
    const unsigned short* __restrict__ Qws,
    const unsigned short* __restrict__ Kws,
    const unsigned short* __restrict__ Vws,
    const float* __restrict__ gamma,
    float* __restrict__ out) {
  __shared__ unsigned short ldsP[8][16 * 64];  // 8 waves x 2KB
  const int lane  = threadIdx.x & 31;
  const int wave  = threadIdx.x >> 5;
  const int w     = blockIdx.x * 8 + wave;
  const int cq    = w & 3;
  const int itile = (w >> 2) & 255;
  const int b     = w >> 10;
  const int i0    = itile * 16;
  const int mrow  = lane & 15;
  const int hi    = (lane >= 16) ? 1 : 0;
  const int cb    = cq * 64;

  // Resident Q A-operand (16 i x 32 d): two b128 loads
  const BfOp qa = load_op_b128(Qws + ((size_t)b * NN + i0 + mrow) * DRR, hi);

  float mstat[8], lstat[8];
  f32x8 acc[4];
  const f32x8 zero = {};
  #pragma unroll
  for (int r = 0; r < 8; ++r) { mstat[r] = -3.0e38f; lstat[r] = 0.f; }
  #pragma unroll
  for (int ct = 0; ct < 4; ++ct) acc[ct] = zero;

  const float LOG2E = 1.44269504088896340736f;
  unsigned short* myP = ldsP[wave];

  for (int j0 = 0; j0 < NN; j0 += 64) {
    // S = Q^T K : four 16x16 tiles (one WMMA each, K-dim = 32)
    f32x8 s[4];
    #pragma unroll
    for (int t = 0; t < 4; ++t) {
      int j = j0 + t * 16 + mrow;
      BfOp kbop = load_op_b128(Kws + ((size_t)b * NN + j) * DRR, hi);
      s[t] = __builtin_amdgcn_wmma_f32_16x16x32_bf16(
          false, qa.v, false, kbop.v, (short)0, zero, false, false);
    }

    // Online softmax. Row of (VGPR r, half) = i0 + r + 8*hi; its 16 columns
    // live in the lanes of this half -> shfl_xor masks 1..8 reduce the row.
    #pragma unroll
    for (int r = 0; r < 8; ++r) {
      float tm = fmaxf(fmaxf(s[0][r], s[1][r]), fmaxf(s[2][r], s[3][r]));
      #pragma unroll
      for (int msk = 1; msk < 16; msk <<= 1) tm = fmaxf(tm, __shfl_xor(tm, msk, 32));
      float mnew  = fmaxf(mstat[r], tm);
      float scale = __builtin_amdgcn_exp2f((mstat[r] - mnew) * LOG2E);
      float p0 = __builtin_amdgcn_exp2f((s[0][r] - mnew) * LOG2E);
      float p1 = __builtin_amdgcn_exp2f((s[1][r] - mnew) * LOG2E);
      float p2 = __builtin_amdgcn_exp2f((s[2][r] - mnew) * LOG2E);
      float p3 = __builtin_amdgcn_exp2f((s[3][r] - mnew) * LOG2E);
      float rs = (p0 + p1) + (p2 + p3);
      #pragma unroll
      for (int msk = 1; msk < 16; msk <<= 1) rs += __shfl_xor(rs, msk, 32);
      lstat[r] = lstat[r] * scale + rs;
      mstat[r] = mnew;
      #pragma unroll
      for (int ct = 0; ct < 4; ++ct) acc[ct][r] *= scale;
      // stash P panel (C/D layout -> row-major 16x64 in LDS)
      unsigned pk01 = pack2bf(p0, p1);
      unsigned pk23 = pack2bf(p2, p3);
      int Mr = r + 8 * hi;
      myP[Mr * 64 + mrow]      = (unsigned short)pk01;
      myP[Mr * 64 + 16 + mrow] = (unsigned short)(pk01 >> 16);
      myP[Mr * 64 + 32 + mrow] = (unsigned short)pk23;
      myP[Mr * 64 + 48 + mrow] = (unsigned short)(pk23 >> 16);
    }

    // Reload P as two A-operands (wave-private LDS; dscnt ordering in-wave)
    BfOp pa[2];
    #pragma unroll
    for (int h = 0; h < 2; ++h)
      pa[h] = load_op_b128(myP + mrow * 64 + h * 32, hi);

    // O += P * V^T over this wave's 4 channel sub-tiles, two k-steps each
    #pragma unroll
    for (int ct = 0; ct < 4; ++ct) {
      int c = cb + ct * 16 + mrow;
      const unsigned short* vrow = Vws + ((size_t)b * CC + c) * NN + j0;
      #pragma unroll
      for (int h = 0; h < 2; ++h) {
        BfOp vb = load_op_b128(vrow + h * 32, hi);
        acc[ct] = __builtin_amdgcn_wmma_f32_16x16x32_bf16(
            false, pa[h].v, false, vb.v, (short)0, acc[ct], false, false);
      }
    }

    // prefetch next K / V blocks (emits global_prefetch_b8)
    if (j0 + 64 < NN) {
      __builtin_prefetch(Kws + ((size_t)b * NN + j0 + 64 + mrow) * DRR, 0, 0);
      __builtin_prefetch(Vws + ((size_t)b * CC + cb + mrow) * NN + j0 + 64, 0, 0);
    }
  }

  const float g = gamma[0];
  #pragma unroll
  for (int r = 0; r < 8; ++r) lstat[r] = __builtin_amdgcn_rcpf(lstat[r]);
  #pragma unroll
  for (int ct = 0; ct < 4; ++ct) {
    int c = cb + ct * 16 + mrow;
    #pragma unroll
    for (int r = 0; r < 8; ++r) {
      int i = i0 + r + 8 * hi;
      size_t idx = ((size_t)b * CC + c) * NN + i;
      out[idx] = g * (acc[ct][r] * lstat[r]) + x[idx];
    }
  }
}

// ---------------------------------------------------------------------------
extern "C" void kernel_launch(void* const* d_in, const int* in_sizes, int n_in,
                              void* d_out, int out_size, void* d_ws, size_t ws_size,
                              hipStream_t stream) {
  const float* x     = (const float*)d_in[0];
  const float* Wq    = (const float*)d_in[1];
  const float* bq    = (const float*)d_in[2];
  const float* Wk    = (const float*)d_in[3];
  const float* bk    = (const float*)d_in[4];
  const float* Wv    = (const float*)d_in[5];
  const float* bv    = (const float*)d_in[6];
  const float* gamma = (const float*)d_in[7];
  float* out = (float*)d_out;

  // workspace: Q [B,N,32] bf16 (2MB) | K (2MB) | V [B,C,N] bf16 (16MB)
  unsigned short* Qws = (unsigned short*)d_ws;
  unsigned short* Kws = Qws + (size_t)BB * NN * DRR;
  unsigned short* Vws = Kws + (size_t)BB * NN * DRR;

  // 2048 waves, 4 per block
  qkv_proj_kernel<<<512, 128, 0, stream>>>(x, Wq, bq, Wk, bk, Wv, bv, Qws, Kws, Vws);
  // 8192 waves, 8 per block
  attn_kernel<<<1024, 256, 0, stream>>>(x, Qws, Kws, Vws, gamma, out);
}
